// RouteNet_Fermi_34462817583671
// MI455X (gfx1250) — matrix operand order repaired and checked by reading, not compiled, not run on hardware
//
#include <hip/hip_runtime.h>
#include <hip/hip_bf16.h>
#include <math.h>

// RouteNet-Fermi on MI455X (gfx1250, wave32).
// GRU gate GEMMs on v_wmma_f32_16x16x32_f16 (f32 accum), fp32 states in
// L2-resident buffers, LDS staging for h C-layout -> A-layout transpose.
// v2: weight B-fragments staged in LDS shared by 4 waves/block, z/r gate
// WMMAs chained, global_prefetch of next-hop state rows.
// v3: sigmoid/tanh/readout reciprocal via v_rcp_f32 (__builtin_amdgcn_rcpf)
// instead of IEEE divide chains (v_div_scale/v_div_fmas) -- cuts the VALU
// count of the gate math by ~2x.

#define HOPS 8
#define DEGQ 64
#define SDIM 32
#define TPB 4   // tiles (waves) per block in path_scan

typedef __attribute__((ext_vector_type(16))) _Float16 v16h;
typedef __attribute__((ext_vector_type(8)))  float    v8f;

__device__ __forceinline__ v8f wmma16(v16h a, v16h b, v8f c) {
  return __builtin_amdgcn_wmma_f32_16x16x32_f16(
      /*neg_a=*/false, a, /*neg_b=*/false, b,
      /*c_mod=*/(short)0, c, /*reuse_a=*/false, /*reuse_b=*/false);
}
__device__ __forceinline__ float rcp_f(float x) { return __builtin_amdgcn_rcpf(x); }
__device__ __forceinline__ float sigm_f(float x) { return rcp_f(1.0f + __expf(-x)); }
__device__ __forceinline__ float tanh_f(float x) { return 1.0f - 2.0f * rcp_f(__expf(2.0f * x) + 1.0f); }
// 16-bit A-matrix 16x32 lane layout (ISA 7.12.2): lane -> M = lane%16,
// half j with g = lane/16 -> K = (j&7) + 8*g + 16*(j>>3)
__device__ __forceinline__ int kmapA(int j, int g) { return (j & 7) + 8 * g + 16 * (j >> 3); }

// ---------------------------------------------------------------------------
// Pack a KxN (N=96) fp32 row-major weight into f16 B-fragments.
// B-fragment (32x16): lane l holds N = nt*16 + l%16, half j holds K = kmapA(j, l/16).
// Fragment order: [kt][nt], kt = K/32 tiles, nt = 6 N tiles. 512 halves each.
// ---------------------------------------------------------------------------
__global__ void pack_weights(const float* __restrict__ W, int K, _Float16* __restrict__ dst) {
  int t = blockIdx.x * blockDim.x + threadIdx.x;
  int nfrag = (K / 32) * 6;
  if (t >= nfrag * 32) return;
  int frag = t >> 5, lane = t & 31;
  int kt = frag / 6, nt = frag % 6;
  int nn = nt * 16 + (lane & 15), g = lane >> 4;
  _Float16* o = dst + (size_t)(frag * 32 + lane) * 16;
  for (int j = 0; j < 16; ++j) o[j] = (_Float16)W[(kt * 32 + kmapA(j, g)) * 96 + nn];
}

// ---------------------------------------------------------------------------
// Embedding kernels (tiny MLPs, scalar VALU)
// ---------------------------------------------------------------------------
__global__ void path_emb(const float* __restrict__ traffic, const float* __restrict__ packets,
                         const int* __restrict__ dist, const float* __restrict__ lam,
                         const float* __restrict__ onb, const float* __restrict__ ont,
                         const float* __restrict__ offt,
                         const float* __restrict__ w1, const float* __restrict__ b1,
                         const float* __restrict__ w2, const float* __restrict__ b2,
                         float* __restrict__ pstate, int P) {
  int p = blockIdx.x * blockDim.x + threadIdx.x;
  if (p >= P) return;
  float f[9];
  f[0] = traffic[p]; f[1] = packets[p];
  int d = dist[p];
  f[2] = (d == 0) ? 1.f : 0.f; f[3] = (d == 1) ? 1.f : 0.f; f[4] = (d == 2) ? 1.f : 0.f;
  f[5] = lam[p]; f[6] = onb[p]; f[7] = ont[p]; f[8] = offt[p];
  float h1[SDIM];
  for (int o = 0; o < SDIM; ++o) {
    float a = b1[o];
    for (int k = 0; k < 9; ++k) a += f[k] * w1[k * SDIM + o];
    h1[o] = fmaxf(a, 0.f);
  }
  for (int o = 0; o < SDIM; ++o) {
    float a = b2[o];
    for (int k = 0; k < SDIM; ++k) a += h1[k] * w2[k * SDIM + o];
    pstate[(size_t)p * SDIM + o] = fmaxf(a, 0.f);
  }
}

__global__ void link_emb(const float* __restrict__ traffic, const float* __restrict__ cap,
                         const int* __restrict__ ptl,
                         const float* __restrict__ w1, const float* __restrict__ b1,
                         const float* __restrict__ w2, const float* __restrict__ b2,
                         float* __restrict__ lstate, int L) {
  int l = blockIdx.x * blockDim.x + threadIdx.x;
  if (l >= L) return;
  float s = 0.f;
  for (int d = 0; d < DEGQ; ++d) s += traffic[ptl[((size_t)l * DEGQ + d) * 2]];
  float load = s / cap[l];
  float h1[SDIM];
  for (int o = 0; o < SDIM; ++o) h1[o] = fmaxf(load * w1[o] + b1[o], 0.f);
  for (int o = 0; o < SDIM; ++o) {
    float a = b2[o];
    for (int k = 0; k < SDIM; ++k) a += h1[k] * w2[k * SDIM + o];
    lstate[(size_t)l * SDIM + o] = fmaxf(a, 0.f);
  }
}

__global__ void queue_emb(const int* __restrict__ btype,
                          const float* __restrict__ w1, const float* __restrict__ b1,
                          const float* __restrict__ w2, const float* __restrict__ b2,
                          float* __restrict__ qstate, int L) {
  int l = blockIdx.x * blockDim.x + threadIdx.x;
  if (l >= L) return;
  int bt = btype[l];
  float h1[SDIM];
  for (int o = 0; o < SDIM; ++o) h1[o] = fmaxf(w1[bt * SDIM + o] + b1[o], 0.f);
  for (int o = 0; o < SDIM; ++o) {
    float a = b2[o];
    for (int k = 0; k < SDIM; ++k) a += h1[k] * w2[k * SDIM + o];
    qstate[(size_t)l * SDIM + o] = fmaxf(a, 0.f);
  }
}

// ---------------------------------------------------------------------------
// Path GRU scan: 4 waves/block, one 16-path tile per wave; 8 steps;
// 18 WMMAs per wave-step. The 18 weight B-fragments (path_Wx 12 + path_Wh 6,
// contiguous) are staged once into LDS and streamed per-WMMA (ds_load_b128),
// keeping wave VGPR usage low. z/r gates accumulate x*Wx and h*Wh into the
// same C fragment; only the n-gate keeps h*Wh separate (n = tanh(xn + r*hn)).
// ---------------------------------------------------------------------------
__global__ void __launch_bounds__(32 * TPB)
path_scan(const float* __restrict__ qstate, const float* __restrict__ lstate,
          const int* __restrict__ ltp,
          const _Float16* __restrict__ Wall,   // 18 frags: Wx[kt=0]x6, Wx[kt=1]x6, Wh x6
          const float* __restrict__ bi, const float* __restrict__ bh,
          const float* __restrict__ h_in, float* __restrict__ h_out,
          float* __restrict__ seq, int P) {
  __shared__ _Float16 wlds[18 * 512];
  __shared__ float hlds[TPB][16 * SDIM];
  const int tid = threadIdx.x;
  const int wave = tid >> 5, lane = tid & 31;
  const int g = lane >> 4, m = lane & 15;
  const int tile = blockIdx.x * TPB + wave;

  // Cooperative weight stage: 18*512 halves = 4608 dwords.
  {
    unsigned int* dst = (unsigned int*)wlds;
    const unsigned int* src = (const unsigned int*)Wall;
    for (int i = tid; i < 18 * 256; i += 32 * TPB) dst[i] = src[i];
  }
  // Per-lane gate biases (z/r: bi+bh folded; n: separate).
  float bzr[4], bxn[2], bhn[2];
  for (int nt = 0; nt < 4; ++nt) bzr[nt] = bi[nt * 16 + m] + bh[nt * 16 + m];
  for (int nt = 0; nt < 2; ++nt) {
    bxn[nt] = bi[(4 + nt) * 16 + m];
    bhn[nt] = bh[(4 + nt) * 16 + m];
  }
  // h0 = path_state tile.
  float* hw = hlds[wave];
  for (int idx = lane; idx < 16 * SDIM; idx += 32) {
    int p = tile * 16 + (idx >> 5);
    hw[idx] = h_in[(size_t)min(p, P - 1) * SDIM + (idx & 31)];
  }
  __syncthreads();

  const int prow = min(tile * 16 + m, P - 1);
  for (int step = 0; step < HOPS; ++step) {
    int link = ltp[(size_t)prow * HOPS + step];
    if (step + 1 < HOPS) {           // prefetch next hop's state rows into WGP$
      int nl = ltp[(size_t)prow * HOPS + step + 1];
      __builtin_prefetch(qstate + (size_t)nl * SDIM, 0, 0);
      __builtin_prefetch(lstate + (size_t)nl * SDIM, 0, 0);
    }
    const float* qs = qstate + (size_t)link * SDIM;
    const float* ls = lstate + (size_t)link * SDIM;
    v16h ax0, ax1, ah;
    for (int j = 0; j < 16; ++j) {
      int K = kmapA(j, g);
      ax0[j] = (_Float16)qs[K];          // features 0..31  (queue)
      ax1[j] = (_Float16)ls[K];          // features 32..63 (link)
      ah[j]  = (_Float16)hw[m * SDIM + K];
    }
    v8f acc[6], hn[2];
    for (int nt = 0; nt < 4; ++nt) {     // z, r gates: fully chained x+h
      v8f c; for (int r = 0; r < 8; ++r) c[r] = bzr[nt];
      c = wmma16(ax0, *(const v16h*)(wlds + (size_t)(nt * 32 + lane) * 16), c);
      c = wmma16(ax1, *(const v16h*)(wlds + (size_t)((6 + nt) * 32 + lane) * 16), c);
      c = wmma16(ah,  *(const v16h*)(wlds + (size_t)((12 + nt) * 32 + lane) * 16), c);
      acc[nt] = c;
    }
    for (int nt = 4; nt < 6; ++nt) {     // n gate: x-part and h-part separate
      v8f c; for (int r = 0; r < 8; ++r) c[r] = bxn[nt - 4];
      c = wmma16(ax0, *(const v16h*)(wlds + (size_t)(nt * 32 + lane) * 16), c);
      c = wmma16(ax1, *(const v16h*)(wlds + (size_t)((6 + nt) * 32 + lane) * 16), c);
      acc[nt] = c;
      v8f d; for (int r = 0; r < 8; ++r) d[r] = bhn[nt - 4];
      hn[nt - 4] = wmma16(ah, *(const v16h*)(wlds + (size_t)((12 + nt) * 32 + lane) * 16), d);
    }
    __syncthreads();
    for (int c2 = 0; c2 < 2; ++c2) {
      for (int r = 0; r < 8; ++r) {
        float z  = sigm_f(acc[c2][r]);
        float rr = sigm_f(acc[2 + c2][r]);
        float nn = tanh_f(acc[4 + c2][r] + rr * hn[c2][r]);
        int row = r + 8 * g, col = c2 * 16 + m;
        float hold = hw[row * SDIM + col];
        float hnew = z * hold + (1.f - z) * nn;
        hw[row * SDIM + col] = hnew;
        int gp = tile * 16 + row;
        if (gp < P) seq[((size_t)gp * HOPS + step) * SDIM + col] = hnew;
      }
    }
    __syncthreads();
  }
  for (int idx = lane; idx < 16 * SDIM; idx += 32) {
    int p = tile * 16 + (idx >> 5);
    if (p < P) h_out[(size_t)p * SDIM + (idx & 31)] = hw[idx];
  }
}

// ---------------------------------------------------------------------------
// Shared wave-level GRU step, x and h both 16x32 fp32 tiles in LDS (K=32).
// 12 WMMAs; z/r gates chained. Writes the updated h back into hlds.
// ---------------------------------------------------------------------------
__device__ void gru_step32(const float* xlds, float* hlds,
                           const _Float16* WxB, const _Float16* WhB,
                           const float* bi, const float* bh, int lane) {
  const int g = lane >> 4, m = lane & 15;
  v16h ax, ah;
  for (int j = 0; j < 16; ++j) {
    int K = kmapA(j, g);
    ax[j] = (_Float16)xlds[m * SDIM + K];
    ah[j] = (_Float16)hlds[m * SDIM + K];
  }
  v8f acc[6], hn[2];
  for (int nt = 0; nt < 4; ++nt) {
    float b = bi[nt * 16 + m] + bh[nt * 16 + m];
    v8f c; for (int r = 0; r < 8; ++r) c[r] = b;
    c = wmma16(ax, *(const v16h*)(WxB + (size_t)(nt * 32 + lane) * 16), c);
    c = wmma16(ah, *(const v16h*)(WhB + (size_t)(nt * 32 + lane) * 16), c);
    acc[nt] = c;
  }
  for (int nt = 4; nt < 6; ++nt) {
    v8f c; for (int r = 0; r < 8; ++r) c[r] = bi[nt * 16 + m];
    acc[nt] = wmma16(ax, *(const v16h*)(WxB + (size_t)(nt * 32 + lane) * 16), c);
    v8f d; for (int r = 0; r < 8; ++r) d[r] = bh[nt * 16 + m];
    hn[nt - 4] = wmma16(ah, *(const v16h*)(WhB + (size_t)(nt * 32 + lane) * 16), d);
  }
  __syncthreads();
  for (int c2 = 0; c2 < 2; ++c2) {
    for (int r = 0; r < 8; ++r) {
      float z  = sigm_f(acc[c2][r]);
      float rr = sigm_f(acc[2 + c2][r]);
      float nn = tanh_f(acc[4 + c2][r] + rr * hn[c2][r]);
      int row = r + 8 * g, col = c2 * 16 + m;
      float hold = hlds[row * SDIM + col];
      hlds[row * SDIM + col] = z * hold + (1.f - z) * nn;
    }
  }
}

// Queue GRU: x = sum over 64 (path,pos) gathers from pss = [prev | seq].
__global__ void __launch_bounds__(32)
queue_gru(const float* __restrict__ pstate_prev, const float* __restrict__ seq,
          const int* __restrict__ ptl,
          const _Float16* __restrict__ WxB, const _Float16* __restrict__ WhB,
          const float* __restrict__ bi, const float* __restrict__ bh,
          float* __restrict__ qstate, int L) {
  __shared__ float xlds[16 * SDIM];
  __shared__ float hlds[16 * SDIM];
  const int tile = blockIdx.x;
  const int lane = threadIdx.x;
  for (int li = 0; li < 16; ++li) {
    int l = min(tile * 16 + li, L - 1);
    float acc = 0.f;
    for (int d = 0; d < DEGQ; ++d) {
      int path = ptl[((size_t)l * DEGQ + d) * 2 + 0];
      int pos  = ptl[((size_t)l * DEGQ + d) * 2 + 1];   // 0..HOPS (0 = prev)
      const float* src = (pos == 0) ? (pstate_prev + (size_t)path * SDIM)
                                    : (seq + ((size_t)path * HOPS + (pos - 1)) * SDIM);
      acc += src[lane];
    }
    xlds[li * SDIM + lane] = acc;
  }
  for (int idx = lane; idx < 16 * SDIM; idx += 32) {
    int l = min(tile * 16 + (idx >> 5), L - 1);
    hlds[idx] = qstate[(size_t)l * SDIM + (idx & 31)];
  }
  __syncthreads();
  gru_step32(xlds, hlds, WxB, WhB, bi, bh, lane);
  __syncthreads();
  for (int idx = lane; idx < 16 * SDIM; idx += 32) {
    int l = tile * 16 + (idx >> 5);
    if (l < L) qstate[(size_t)l * SDIM + (idx & 31)] = hlds[idx];
  }
}

// Link GRU: x = queue_state[queue_to_link[l]].
__global__ void __launch_bounds__(32)
link_gru(const float* __restrict__ qstate, const int* __restrict__ q2l,
         const _Float16* __restrict__ WxB, const _Float16* __restrict__ WhB,
         const float* __restrict__ bi, const float* __restrict__ bh,
         float* __restrict__ lstate, int L) {
  __shared__ float xlds[16 * SDIM];
  __shared__ float hlds[16 * SDIM];
  const int tile = blockIdx.x;
  const int lane = threadIdx.x;
  for (int li = 0; li < 16; ++li) {
    int l = min(tile * 16 + li, L - 1);
    xlds[li * SDIM + lane] = qstate[(size_t)q2l[l] * SDIM + lane];
  }
  for (int idx = lane; idx < 16 * SDIM; idx += 32) {
    int l = min(tile * 16 + (idx >> 5), L - 1);
    hlds[idx] = lstate[(size_t)l * SDIM + (idx & 31)];
  }
  __syncthreads();
  gru_step32(xlds, hlds, WxB, WhB, bi, bh, lane);
  __syncthreads();
  for (int idx = lane; idx < 16 * SDIM; idx += 32) {
    int l = tile * 16 + (idx >> 5);
    if (l < L) lstate[(size_t)l * SDIM + (idx & 31)] = hlds[idx];
  }
}

// ---------------------------------------------------------------------------
// Readout: per path, 8 hops of 32->16->16->1 MLP + delay accumulation.
// ---------------------------------------------------------------------------
__global__ void readout(const float* __restrict__ seq, const int* __restrict__ ltp,
                        const float* __restrict__ cap,
                        const float* __restrict__ w1, const float* __restrict__ b1,
                        const float* __restrict__ w2, const float* __restrict__ b2,
                        const float* __restrict__ w3, const float* __restrict__ b3,
                        const float* __restrict__ traffic, const float* __restrict__ packets,
                        float* __restrict__ out, int P) {
  int p = blockIdx.x * blockDim.x + threadIdx.x;
  if (p >= P) return;
  float qsum = 0.f, csum = 0.f;
  for (int t = 0; t < HOPS; ++t) {
    int link = ltp[(size_t)p * HOPS + t];
    float rcg = rcp_f(cap[link] * 1.0e9f);   // one v_rcp serves both delays
    const float* hv = seq + ((size_t)p * HOPS + t) * SDIM;
    float h1[16], h2[16];
    for (int o = 0; o < 16; ++o) {
      float a = b1[o];
      for (int k = 0; k < SDIM; ++k) a += hv[k] * w1[k * 16 + o];
      h1[o] = fmaxf(a, 0.f);
    }
    for (int o = 0; o < 16; ++o) {
      float a = b2[o];
      for (int k = 0; k < 16; ++k) a += h1[k] * w2[k * 16 + o];
      h2[o] = fmaxf(a, 0.f);
    }
    float occ = b3[0];
    for (int k = 0; k < 16; ++k) occ += h2[k] * w3[k];
    qsum += occ * rcg;
    csum += rcg;
  }
  out[p] = qsum + (traffic[p] / packets[p]) * csum;
}

// ---------------------------------------------------------------------------
extern "C" void kernel_launch(void* const* d_in, const int* in_sizes, int n_in,
                              void* d_out, int out_size, void* d_ws, size_t ws_size,
                              hipStream_t stream) {
  const int P = in_sizes[0];   // 50000
  const int L = in_sizes[6];   // 8000

  const float* traffic = (const float*)d_in[0];
  const float* packets = (const float*)d_in[1];
  const float* lam     = (const float*)d_in[2];
  const float* onb     = (const float*)d_in[3];
  const float* ont     = (const float*)d_in[4];
  const float* offt    = (const float*)d_in[5];
  const float* cap     = (const float*)d_in[6];
  const float* pe_w1 = (const float*)d_in[7],  *pe_b1 = (const float*)d_in[8];
  const float* pe_w2 = (const float*)d_in[9],  *pe_b2 = (const float*)d_in[10];
  const float* le_w1 = (const float*)d_in[11], *le_b1 = (const float*)d_in[12];
  const float* le_w2 = (const float*)d_in[13], *le_b2 = (const float*)d_in[14];
  const float* qe_w1 = (const float*)d_in[15], *qe_b1 = (const float*)d_in[16];
  const float* qe_w2 = (const float*)d_in[17], *qe_b2 = (const float*)d_in[18];
  const float* pWx = (const float*)d_in[19], *pWh = (const float*)d_in[20];
  const float* pbi = (const float*)d_in[21], *pbh = (const float*)d_in[22];
  const float* qWx = (const float*)d_in[23], *qWh = (const float*)d_in[24];
  const float* qbi = (const float*)d_in[25], *qbh = (const float*)d_in[26];
  const float* lWx = (const float*)d_in[27], *lWh = (const float*)d_in[28];
  const float* lbi = (const float*)d_in[29], *lbh = (const float*)d_in[30];
  const float* ro_w1 = (const float*)d_in[31], *ro_b1 = (const float*)d_in[32];
  const float* ro_w2 = (const float*)d_in[33], *ro_b2 = (const float*)d_in[34];
  const float* ro_w3 = (const float*)d_in[35], *ro_b3 = (const float*)d_in[36];
  const int* dist  = (const int*)d_in[37];
  const int* btype = (const int*)d_in[38];
  const int* ltp   = (const int*)d_in[39];
  const int* ptl   = (const int*)d_in[40];
  const int* q2l   = (const int*)d_in[41];

  // Workspace carve (all 256B aligned): ~66 MB total.
  char* w = (char*)d_ws;
  auto carve = [&](size_t bytes) { char* p = w; w += (bytes + 255) & ~(size_t)255; return p; };
  float* pstateA = (float*)carve((size_t)P * SDIM * 4);
  float* pstateB = (float*)carve((size_t)P * SDIM * 4);
  float* qstate  = (float*)carve((size_t)L * SDIM * 4);
  float* lstate  = (float*)carve((size_t)L * SDIM * 4);
  float* seq     = (float*)carve((size_t)P * HOPS * SDIM * 4);
  _Float16* wpack = (_Float16*)carve((size_t)42 * 512 * 2);
  _Float16* wp_pWx = wpack + 0 * 512;    // 12 frags (K=64), then Wh: contiguous
  _Float16* wp_pWh = wpack + 12 * 512;   // 6 frags
  _Float16* wp_qWx = wpack + 18 * 512;
  _Float16* wp_qWh = wpack + 24 * 512;
  _Float16* wp_lWx = wpack + 30 * 512;
  _Float16* wp_lWh = wpack + 36 * 512;

  // Pack weights to B-fragment order (once per launch; cheap).
  pack_weights<<<3, 128, 0, stream>>>(pWx, 64, wp_pWx);
  pack_weights<<<2, 128, 0, stream>>>(pWh, 32, wp_pWh);
  pack_weights<<<2, 128, 0, stream>>>(qWx, 32, wp_qWx);
  pack_weights<<<2, 128, 0, stream>>>(qWh, 32, wp_qWh);
  pack_weights<<<2, 128, 0, stream>>>(lWx, 32, wp_lWx);
  pack_weights<<<2, 128, 0, stream>>>(lWh, 32, wp_lWh);

  // Embeddings.
  path_emb<<<(P + 255) / 256, 256, 0, stream>>>(traffic, packets, dist, lam, onb, ont, offt,
                                                pe_w1, pe_b1, pe_w2, pe_b2, pstateA, P);
  link_emb<<<(L + 255) / 256, 256, 0, stream>>>(traffic, cap, ptl,
                                                le_w1, le_b1, le_w2, le_b2, lstate, L);
  queue_emb<<<(L + 255) / 256, 256, 0, stream>>>(btype, qe_w1, qe_b1, qe_w2, qe_b2, qstate, L);

  // Message-passing iterations.
  float* cur = pstateA; float* nxt = pstateB;
  const int ptiles = (P + 15) / 16;
  const int pblocks = (ptiles + TPB - 1) / TPB;
  const int ltiles = (L + 15) / 16;
  for (int it = 0; it < 8; ++it) {
    path_scan<<<pblocks, 32 * TPB, 0, stream>>>(qstate, lstate, ltp, wp_pWx, pbi, pbh,
                                                cur, nxt, seq, P);
    queue_gru<<<ltiles, 32, 0, stream>>>(cur, seq, ptl, wp_qWx, wp_qWh, qbi, qbh, qstate, L);
    link_gru<<<ltiles, 32, 0, stream>>>(qstate, q2l, wp_lWx, wp_lWh, lbi, lbh, lstate, L);
    float* tmp = cur; cur = nxt; nxt = tmp;
  }

  // Readout.
  readout<<<(P + 255) / 256, 256, 0, stream>>>(seq, ltp, cap, ro_w1, ro_b1, ro_w2, ro_b2,
                                               ro_w3, ro_b3, traffic, packets,
                                               (float*)d_out, P);
}
